// aux_branch_35880156791486
// MI455X (gfx1250) — matrix operand order repaired and checked by reading, not compiled, not run on hardware
//
#include <hip/hip_runtime.h>
#include <hip/hip_bf16.h>

typedef __attribute__((ext_vector_type(16))) _Float16 v16h;
typedef __attribute__((ext_vector_type(8)))  float    v8f;

#define NGROUP 16
#define GPTS   4096
#define EPSBN  1e-5f

struct MlpW { const _Float16* W; const float* sc; const float* sh; };

// ---------------------------------------------------------------------------
// WMMA pointwise MLP layer:  out = relu(scale * (in @ W) + shift), f16 out.
// in : LDS, rows x kpad halves (kpad % 32 == 0, zero padded)
// W  : global, pre-swizzled FRAGMENT-MAJOR layout (see prep_weights_kernel):
//      for k-chunk kc, n-tile nt, lane l: the 16-half B fragment is stored
//      contiguously at ((kc/32 * nt_n + nt) * 32 + l) * 16, 32B aligned.
//      => one v16h load (2x global_load_b128) per fragment instead of 16
//      strided 2-byte loads.
// All waves of the block cooperate over (row-tile, col-tile) tasks.
// EXEC is all-ones on every path reaching the WMMA (uniform control flow).
// ---------------------------------------------------------------------------
__device__ __forceinline__ void mlp_layer_wmma(
    MlpW L, const _Float16* __restrict__ in, _Float16* __restrict__ out,
    int rows, int kpad, int cout)
{
  const int wave = threadIdx.x >> 5;
  const int nw   = blockDim.x >> 5;
  const int lane = threadIdx.x & 31;
  const int hi   = lane >> 4;      // half-wave select
  const int ln   = lane & 15;
  const int mt_n = rows >> 4, nt_n = cout >> 4;
  const size_t fstride = (size_t)nt_n * 512;   // halves per K-chunk of fragments
  for (int t = wave; t < mt_n * nt_n; t += nw) {
    const int mt = t / nt_n, nt = t - mt * nt_n;
    const int col = nt * 16 + ln;
    const _Float16* bp = L.W + ((size_t)nt * 32 + lane) * 16;
    v8f acc = {};
    for (int kc = 0; kc < kpad; kc += 32) {
      // A fragment (16x32 f16): lane<16 owns K kc+0..7 / kc+16..23,
      // lane>=16 owns K kc+8..15 / kc+24..31 for row mt*16+ln.
      v16h a;
      const _Float16* ap = in + (mt * 16 + ln) * kpad + kc + hi * 8;
      #pragma unroll
      for (int i = 0; i < 8; ++i) { a[i] = ap[i]; a[i + 8] = ap[i + 16]; }
      // B fragment: single 32-byte vector load from fragment-major weights.
      v16h b = *(const v16h*)bp;
      __builtin_prefetch(bp + fstride, 0, 0);  // global_prefetch_b8 next K-chunk
      bp += fstride;
      acc = __builtin_amdgcn_wmma_f32_16x16x32_f16(false, a, false, b,
                                                   (short)0, acc, false, false);
    }
    // C/D layout: lane<16 -> rows mt*16+0..7, lane>=16 -> rows mt*16+8..15
    const float sc = L.sc[col], sh = L.sh[col];
    const int mb = mt * 16 + hi * 8;
    #pragma unroll
    for (int j = 0; j < 8; ++j) {
      float v = acc[j] * sc + sh;
      v = v > 0.f ? v : 0.f;
      out[(mb + j) * cout + col] = (_Float16)v;
    }
  }
}

// ---------------------------------------------------------------------------
// Weight prep: f32 -> f16 with K zero-padding, swizzled to WMMA fragment-
// major layout; fold BN into scale/shift.
//   y = ((x@W + b) - mu) * rsqrt(var+eps) * g + be  =  acc*sc + sh
// Fragment layout (ISA 16-bit B-matrix 32x16): lane l holds
//   b[i] = W[kc*32 + (l>>4)*8 + (i&7) + (i>>3)*16][nt*16 + (l&15)],  i = 0..15
// ---------------------------------------------------------------------------
struct WPrepEntry {
  const float *W, *b, *g, *be, *mu, *var;
  _Float16* Wh; float* sc; float* sh;
  int cin, kpad, cout;
};
struct WPrepAll { WPrepEntry e[14]; };

__global__ void prep_weights_kernel(WPrepAll p) {
  WPrepEntry E = p.e[blockIdx.x];
  const int nt_n = E.cout >> 4;
  const int n = E.kpad * E.cout;
  for (int t = threadIdx.x; t < n; t += blockDim.x) {
    const int i    = t & 15;          // element within fragment
    const int frag = t >> 4;
    const int lane = frag & 31;
    const int tile = frag >> 5;
    const int nt   = tile % nt_n;
    const int kc   = tile / nt_n;
    const int k    = kc * 32 + (lane >> 4) * 8 + (i & 7) + (i >> 3) * 16;
    const int c    = nt * 16 + (lane & 15);
    E.Wh[t] = (_Float16)(k < E.cin ? E.W[(size_t)k * E.cout + c] : 0.f);
  }
  for (int c = threadIdx.x; c < E.cout; c += blockDim.x) {
    float s = E.g[c] * rsqrtf(E.var[c] + EPSBN);
    E.sc[c] = s;
    E.sh[c] = (E.b[c] - E.mu[c]) * s + E.be[c];
  }
}

// ---------------------------------------------------------------------------
// Reshape (B,6,N) channel-major points into per-group xyz / 6-ch features.
// Labels are arange//M + 1 => argsort is identity, groups = flat reshape.
// ---------------------------------------------------------------------------
__global__ void group_points_kernel(const float* __restrict__ pts,
                                    float* __restrict__ xyz0,
                                    float* __restrict__ feat0) {
  int r = blockIdx.x * blockDim.x + threadIdx.x;   // 65536 flat rows
  if (r >= NGROUP * GPTS) return;
  int b = r >> 14, n = r & 16383;
  #pragma unroll
  for (int c = 0; c < 6; ++c) {
    float v = pts[((size_t)b * 6 + c) * 16384 + n];
    feat0[(size_t)r * 6 + c] = v;
    if (c < 3) xyz0[(size_t)r * 3 + c] = v;
  }
}

// ---------------------------------------------------------------------------
// Farthest point sampling: one block per group, LDS dist + argmax reduction.
// Tie-break = lowest index (matches jnp.argmax first-max semantics).
// Emits new_xyz[it] = xyz[far] with far emitted BEFORE the dist update.
// ---------------------------------------------------------------------------
template<int N, int NP>
__global__ void __launch_bounds__(256) fps_kernel(const float* __restrict__ xyz,
                                                  float* __restrict__ new_xyz) {
  __shared__ float dist[N];
  __shared__ float rv[256];
  __shared__ int   ri[256];
  __shared__ float fp[3];
  __shared__ int   farS;
  const int tid = threadIdx.x;
  const float* xg = xyz + (size_t)blockIdx.x * N * 3;
  float* nx = new_xyz + (size_t)blockIdx.x * NP * 3;
  for (int i = tid; i < N; i += 256) dist[i] = 1e10f;
  if (tid == 0) farS = 0;
  __syncthreads();
  for (int it = 0; it < NP; ++it) {
    if (tid == 0) {
      int f = farS;
      fp[0] = xg[f * 3 + 0]; fp[1] = xg[f * 3 + 1]; fp[2] = xg[f * 3 + 2];
      nx[it * 3 + 0] = fp[0]; nx[it * 3 + 1] = fp[1]; nx[it * 3 + 2] = fp[2];
    }
    __syncthreads();
    const float fx = fp[0], fy = fp[1], fz = fp[2];
    float bm = -1.f; int bi = 0;
    for (int i = tid; i < N; i += 256) {
      float dx = xg[i * 3] - fx, dy = xg[i * 3 + 1] - fy, dz = xg[i * 3 + 2] - fz;
      float nd = fminf(dist[i], dx * dx + dy * dy + dz * dz);
      dist[i] = nd;
      if (nd > bm) { bm = nd; bi = i; }
    }
    rv[tid] = bm; ri[tid] = bi;
    __syncthreads();
    for (int s = 128; s > 0; s >>= 1) {
      if (tid < s) {
        float ov = rv[tid + s]; int oi = ri[tid + s];
        if (ov > rv[tid] || (ov == rv[tid] && oi < ri[tid])) { rv[tid] = ov; ri[tid] = oi; }
      }
      __syncthreads();
    }
    if (tid == 0) farS = ri[0];
    __syncthreads();
  }
}

// ---------------------------------------------------------------------------
// Ball query: first 32 in-radius indices in ascending order, pad with first.
// ---------------------------------------------------------------------------
template<int N, int NP>
__global__ void ball_query_kernel(const float* __restrict__ xyz,
                                  const float* __restrict__ new_xyz,
                                  int* __restrict__ idx, float r2) {
  int t = blockIdx.x * blockDim.x + threadIdx.x;
  if (t >= NGROUP * NP) return;
  int g = t / NP;
  const float* xg = xyz + (size_t)g * N * 3;
  const float* nq = new_xyz + (size_t)t * 3;
  int* out = idx + (size_t)t * 32;
  const float cx = nq[0], cy = nq[1], cz = nq[2];
  int cnt = 0, first = 0;
  for (int j = 0; j < N && cnt < 32; ++j) {
    float dx = xg[j * 3] - cx, dy = xg[j * 3 + 1] - cy, dz = xg[j * 3 + 2] - cz;
    if (dx * dx + dy * dy + dz * dz <= r2) {
      if (cnt == 0) first = j;
      out[cnt++] = j;
    }
  }
  for (; cnt < 32; ++cnt) out[cnt] = first;
}

// ---------------------------------------------------------------------------
// SA stage: gather 32 neighbors -> concat(rel-xyz, feat) -> 3 WMMA MLP
// layers -> channelwise max over the 32 samples. One block per query point.
// ---------------------------------------------------------------------------
template<int NIN, int NP, int CIN, int KPAD0, int C1, int C2, int C3, int WA, int WB>
__global__ void __launch_bounds__(256) sa_mlp_kernel(
    const float* __restrict__ xyz, const float* __restrict__ feat,
    const float* __restrict__ nxyz, const int* __restrict__ bidx,
    MlpW l0, MlpW l1, MlpW l2, float* __restrict__ pooled) {
  __shared__ _Float16 bufA[32 * WA];   // holds widths {KPAD0, C2}
  __shared__ _Float16 bufB[32 * WB];   // holds widths {C1, C3}
  __shared__ int   sidx[32];
  __shared__ float cxyz[3];
  const int g = blockIdx.y, q = blockIdx.x, tid = threadIdx.x;
  const size_t qp = (size_t)g * NP + q;
  if (tid < 32) sidx[tid] = bidx[qp * 32 + tid];
  if (tid < 3)  cxyz[tid] = nxyz[qp * 3 + tid];
  __syncthreads();
  const float* xg = xyz + (size_t)g * NIN * 3;
  const float* fg = feat + (size_t)g * NIN * CIN;
  for (int e = tid; e < 32 * KPAD0; e += 256) {
    int r = e / KPAD0, c = e - r * KPAD0;
    float v = 0.f;
    int j = sidx[r];
    if (c < 3)            v = xg[(size_t)j * 3 + c] - cxyz[c];
    else if (c < 3 + CIN) v = fg[(size_t)j * CIN + (c - 3)];
    bufA[e] = (_Float16)v;
  }
  __syncthreads();
  mlp_layer_wmma(l0, bufA, bufB, 32, KPAD0, C1); __syncthreads();
  mlp_layer_wmma(l1, bufB, bufA, 32, C1,    C2); __syncthreads();
  mlp_layer_wmma(l2, bufA, bufB, 32, C2,    C3); __syncthreads();
  for (int c = tid; c < C3; c += 256) {
    float m = -1e30f;
    for (int r = 0; r < 32; ++r) m = fmaxf(m, (float)bufB[r * C3 + c]);
    pooled[qp * C3 + c] = m;
  }
}

// ---------------------------------------------------------------------------
// Projection head (512->256->128 WMMA) + per-row L2 normalize; writes
// normalized feats to ws and cur_feat rows (+ class-id col) to d_out.
// One block per group (16 rows).
// ---------------------------------------------------------------------------
__global__ void __launch_bounds__(256) proj_kernel(
    const float* __restrict__ p4, MlpW l0, MlpW l1,
    float* __restrict__ feats, float* __restrict__ curfeat) {
  __shared__ _Float16 bufA[16 * 512];
  __shared__ _Float16 bufB[16 * 256];
  __shared__ float fbuf[16 * 128];
  __shared__ float red[256];
  const int g = blockIdx.x, tid = threadIdx.x;
  const float* src = p4 + (size_t)g * 16 * 512;
  for (int e = tid; e < 16 * 512; e += 256) bufA[e] = (_Float16)src[e];
  __syncthreads();
  mlp_layer_wmma(l0, bufA, bufB, 16, 512, 256); __syncthreads();
  mlp_layer_wmma(l1, bufB, bufA, 16, 256, 128); __syncthreads();
  for (int e = tid; e < 16 * 128; e += 256) fbuf[e] = (float)bufA[e];
  __syncthreads();
  const int r = tid >> 4, c0 = tid & 15;   // 16 rows x 16 threads
  float s = 0.f;
  #pragma unroll
  for (int k = 0; k < 8; ++k) { float v = fbuf[r * 128 + c0 * 8 + k]; s += v * v; }
  red[tid] = s;
  __syncthreads();
  for (int st = 8; st > 0; st >>= 1) {
    if (c0 < st) red[tid] += red[tid + st];
    __syncthreads();
  }
  const float inv = 1.f / fmaxf(sqrtf(red[r * 16]), 1e-12f);
  #pragma unroll
  for (int k = 0; k < 8; ++k) {
    int c = c0 * 8 + k;
    float v = fbuf[r * 128 + c] * inv;
    feats[((size_t)g * 16 + r) * 128 + c] = v;
    curfeat[((size_t)g * 16 + r) * 129 + c] = v;
  }
  if (c0 == 0) curfeat[((size_t)g * 16 + r) * 129 + 128] = (float)(g + 1);
}

// prior_new = beta*prior + (1-beta)*mean(feats over 16 pts)
__global__ void ema_kernel(const float* __restrict__ feats,
                           const float* __restrict__ prior,
                           float* __restrict__ out_prior) {
  const int g = blockIdx.x, c = threadIdx.x;   // 128 threads
  float s = 0.f;
  for (int r = 0; r < 16; ++r) s += feats[((size_t)g * 16 + r) * 128 + c];
  out_prior[g * 128 + c] = 0.999f * prior[g * 128 + c] + 0.001f * (s * (1.f / 16.f));
}

// ---------------------------------------------------------------------------
extern "C" void kernel_launch(void* const* d_in, const int* in_sizes, int n_in,
                              void* d_out, int out_size, void* d_ws, size_t ws_size,
                              hipStream_t stream) {
  (void)in_sizes; (void)n_in; (void)out_size; (void)ws_size;
  // Input leaf order (setup_inputs insertion order):
  // 0 points, 1 label(unused: identity sort), 2 prior_ema0,
  // 3.. params leaves: sa1(3 layers),sa2,sa3,sa4,proj(2) each (W,b,g,be,mu,var),
  // then num_class, level (unused).
  const float* points = (const float*)d_in[0];
  const float* prior  = (const float*)d_in[2];
  auto F = [&](int i) { return (const float*)d_in[i]; };

  struct LCfg { int cin, kpad, cout; };
  static const LCfg lc[14] = {
    {9, 32, 32}, {32, 32, 32}, {32, 32, 64},
    {67, 96, 64}, {64, 64, 64}, {64, 64, 128},
    {131, 160, 128}, {128, 128, 128}, {128, 128, 256},
    {259, 288, 256}, {256, 256, 256}, {256, 256, 512},
    {512, 512, 256}, {256, 256, 128}};
  size_t whOff[14], ssOff[14], hsum = 0, csum = 0;
  for (int i = 0; i < 14; ++i) {
    whOff[i] = hsum; hsum += (size_t)lc[i].kpad * lc[i].cout;
    ssOff[i] = csum; csum += lc[i].cout;
  }

  // Workspace carve-up (everything written before read on this stream).
  char* base = (char*)d_ws;
  size_t off = 0;
  auto take = [&](size_t b) -> char* {
    char* p = base + off; off = (off + b + 255) & ~(size_t)255; return p;
  };
  float* xyz0  = (float*)take((size_t)NGROUP * GPTS * 3 * 4);
  float* feat0 = (float*)take((size_t)NGROUP * GPTS * 6 * 4);
  float* x1 = (float*)take((size_t)NGROUP * 1024 * 3 * 4);
  float* p1 = (float*)take((size_t)NGROUP * 1024 * 64 * 4);
  float* x2 = (float*)take((size_t)NGROUP * 256 * 3 * 4);
  float* p2 = (float*)take((size_t)NGROUP * 256 * 128 * 4);
  float* x3 = (float*)take((size_t)NGROUP * 64 * 3 * 4);
  float* p3 = (float*)take((size_t)NGROUP * 64 * 256 * 4);
  float* x4 = (float*)take((size_t)NGROUP * 16 * 3 * 4);
  float* p4 = (float*)take((size_t)NGROUP * 16 * 512 * 4);
  int* b1 = (int*)take((size_t)NGROUP * 1024 * 32 * 4);
  int* b2 = (int*)take((size_t)NGROUP * 256 * 32 * 4);
  int* b3 = (int*)take((size_t)NGROUP * 64 * 32 * 4);
  int* b4 = (int*)take((size_t)NGROUP * 16 * 32 * 4);
  _Float16* wh   = (_Float16*)take(hsum * 2);
  float*    scA  = (float*)take(csum * 4);
  float*    shA  = (float*)take(csum * 4);
  float*    feats = (float*)take((size_t)256 * 128 * 4);

  WPrepAll wp;
  MlpW mw[14];
  for (int i = 0; i < 14; ++i) {
    const int k = 3 + 6 * i;
    wp.e[i] = {F(k), F(k + 1), F(k + 2), F(k + 3), F(k + 4), F(k + 5),
               wh + whOff[i], scA + ssOff[i], shA + ssOff[i],
               lc[i].cin, lc[i].kpad, lc[i].cout};
    mw[i] = {wh + whOff[i], scA + ssOff[i], shA + ssOff[i]};
  }

  float* out = (float*)d_out;
  float* out_prior = out + 256 * 129;

  prep_weights_kernel<<<14, 256, 0, stream>>>(wp);
  group_points_kernel<<<(NGROUP * GPTS + 255) / 256, 256, 0, stream>>>(points, xyz0, feat0);

  // SA1: 4096 -> 1024, r=0.1, 9 -> 32 -> 32 -> 64
  fps_kernel<4096, 1024><<<NGROUP, 256, 0, stream>>>(xyz0, x1);
  ball_query_kernel<4096, 1024><<<(NGROUP * 1024 + 255) / 256, 256, 0, stream>>>(xyz0, x1, b1, 0.01f);
  sa_mlp_kernel<4096, 1024, 6, 32, 32, 32, 64, 32, 64>
      <<<dim3(1024, NGROUP), 256, 0, stream>>>(xyz0, feat0, x1, b1, mw[0], mw[1], mw[2], p1);

  // SA2: 1024 -> 256, r=0.2, 67 -> 64 -> 64 -> 128
  fps_kernel<1024, 256><<<NGROUP, 256, 0, stream>>>(x1, x2);
  ball_query_kernel<1024, 256><<<(NGROUP * 256 + 255) / 256, 256, 0, stream>>>(x1, x2, b2, 0.04f);
  sa_mlp_kernel<1024, 256, 64, 96, 64, 64, 128, 96, 128>
      <<<dim3(256, NGROUP), 256, 0, stream>>>(x1, p1, x2, b2, mw[3], mw[4], mw[5], p2);

  // SA3: 256 -> 64, r=0.4, 131 -> 128 -> 128 -> 256
  fps_kernel<256, 64><<<NGROUP, 256, 0, stream>>>(x2, x3);
  ball_query_kernel<256, 64><<<(NGROUP * 64 + 255) / 256, 256, 0, stream>>>(x2, x3, b3, 0.16f);
  sa_mlp_kernel<256, 64, 128, 160, 128, 128, 256, 160, 256>
      <<<dim3(64, NGROUP), 256, 0, stream>>>(x2, p2, x3, b3, mw[6], mw[7], mw[8], p3);

  // SA4: 64 -> 16, r=0.8, 259 -> 256 -> 256 -> 512
  fps_kernel<64, 16><<<NGROUP, 256, 0, stream>>>(x3, x4);
  ball_query_kernel<64, 16><<<(NGROUP * 16 + 255) / 256, 256, 0, stream>>>(x3, x4, b4, 0.64f);
  sa_mlp_kernel<64, 16, 256, 288, 256, 256, 512, 288, 512>
      <<<dim3(16, NGROUP), 256, 0, stream>>>(x3, p3, x4, b4, mw[9], mw[10], mw[11], p4);

  // Projection + normalize + cur_feat, then EMA prior.
  proj_kernel<<<NGROUP, 256, 0, stream>>>(p4, mw[12], mw[13], feats, out);
  ema_kernel<<<NGROUP, 128, 0, stream>>>(feats, prior, out_prior);
}
